// ExportCausalSelfAttention_63324997812732
// MI455X (gfx1250) — matrix operand order repaired and checked by reading, not compiled, not run on hardware
//
#include <hip/hip_runtime.h>

typedef __bf16 bf16;
typedef __attribute__((ext_vector_type(16))) __bf16 v16bf;
typedef __attribute__((ext_vector_type(8)))  __bf16 v8bf;
typedef __attribute__((ext_vector_type(4)))  __bf16 v4bf;
typedef __attribute__((ext_vector_type(2)))  __bf16 v2bf;
typedef __attribute__((ext_vector_type(8)))  float  v8f;
typedef __attribute__((ext_vector_type(4)))  float  v4f;
typedef __attribute__((ext_vector_type(4)))  unsigned v4u;
typedef __attribute__((ext_vector_type(8)))  int    v8i;
typedef __attribute__((ext_vector_type(4)))  int    v4i;

#define T_SEQ   2048
#define C_EMB   2048
#define NHEAD   32
#define NKV     4
#define HD      128
#define QDIM    (NHEAD*HD)        // 4096
#define KVDIM   (NKV*HD)          // 512
#define QKV_N   (QDIM + 2*KVDIM)  // 5120

static __device__ inline v8f v8f_zero() {
  v8f z;
#pragma unroll
  for (int i = 0; i < 8; ++i) z[i] = 0.0f;
  return z;
}

static __device__ inline v16bf mk16(v8bf lo, v8bf hi) {
  v16bf r;
#pragma unroll
  for (int i = 0; i < 8; ++i) { r[i] = lo[i]; r[i + 8] = hi[i]; }
  return r;
}

// LDS byte offset of a __shared__ object (generic -> AS3 -> integer)
static __device__ inline unsigned lds_byte_off(const void* p) {
  return (unsigned)(uintptr_t)(__attribute__((address_space(3))) const char*)p;
}

// ---------------------------------------------------------------- TDM tile load
// DMA one 128x32 bf16 tile (rows x k) from global (row stride = strideK elems)
// into LDS at lds_addr, row-major packed [128][32]. D# per ISA 08 sec 8.
static __device__ inline void tdm_load_tile(unsigned lds_addr, const bf16* gsrc,
                                            int strideK) {
  unsigned long long ga = (unsigned long long)(uintptr_t)gsrc;
  v4u g0;
  g0[0] = 1u;                                            // count=1, user mode
  g0[1] = lds_addr;                                      // lds_addr [63:32]
  g0[2] = (unsigned)ga;                                  // global_addr lo
  g0[3] = (unsigned)((ga >> 32) & 0x1FFFFFFull)          // global_addr[56:32]
          | (2u << 30);                                  // type=2 ("image")
  v8i g1;
  g1[0] = (int)(1u << 16);        // wg_mask=0 | data_size=1 (2B) | no flags
  g1[1] = (int)(32u << 16);       // atomic_barrier=0 | tensor_dim0.lo16 = 32
  g1[2] = (int)(128u << 16);      // tensor_dim0.hi16=0 | tensor_dim1.lo16 = 128
  g1[3] = (int)(32u << 16);       // tensor_dim1.hi16=0 | tile_dim0 = 32
  g1[4] = 128;                    // tile_dim1 = 128 | tile_dim2 = 0
  g1[5] = strideK;                // tensor_dim0_stride lo32 (elements)
  g1[6] = 0;                      // stride0.hi16 | stride1.lo16
  g1[7] = 0;
  v4i z4; z4[0] = z4[1] = z4[2] = z4[3] = 0;
#if __has_include(<hip/amd_detail/amd_gfx1250_TDM.h>)
  v8i z8;
#pragma unroll
  for (int i = 0; i < 8; ++i) z8[i] = 0;
  __builtin_amdgcn_tensor_load_to_lds(g0, g1, z4, z4, z8, 0);
#else
  __builtin_amdgcn_tensor_load_to_lds(g0, g1, z4, z4, 0);
#endif
}

// ---------------------------------------------------------------- convert
__global__ __launch_bounds__(256)
void cvt_f32_bf16(const float* __restrict__ src, bf16* __restrict__ dst, int n4) {
  int i = blockIdx.x * blockDim.x + threadIdx.x;
  if (i < n4) {
    v4f x = *(const v4f*)(src + 4 * (size_t)i);
    v4bf y;
    y[0] = (bf16)x[0]; y[1] = (bf16)x[1]; y[2] = (bf16)x[2]; y[3] = (bf16)x[3];
    *(v4bf*)(dst + 4 * (size_t)i) = y;
  }
}

// ---------------------------------------------------------------- GEMM (TDM path)
// C[M,N] (f32) = A[M,K] (bf16, row-major) * B[N,K]^T (bf16, row-major weights)
// 256 threads (8 waves), tile 128x128, BK=32, double-buffered LDS fed by TDM.
__global__ __launch_bounds__(256)
void gemm_bf16_wmma_tdm(const bf16* __restrict__ A, const bf16* __restrict__ B,
                        float* __restrict__ C, int M, int N, int K) {
  __shared__ __align__(16) bf16 lA[2][128 * 32];
  __shared__ __align__(16) bf16 lB[2][128 * 32];

  const int tid  = threadIdx.x;
  const int w    = tid >> 5;
  const int lane = tid & 31;
  const int g    = lane >> 4;
  const int r    = lane & 15;

  const int row0 = blockIdx.y * 128;
  const int col0 = blockIdx.x * 128;
  const int waveR = (w >> 1) * 32;
  const int waveC = (w & 1) * 64;

  unsigned offA[2], offB[2];
  offA[0] = lds_byte_off(&lA[0][0]); offA[1] = lds_byte_off(&lA[1][0]);
  offB[0] = lds_byte_off(&lB[0][0]); offB[1] = lds_byte_off(&lB[1][0]);

  v8f acc[2][4];
#pragma unroll
  for (int mt = 0; mt < 2; ++mt)
#pragma unroll
    for (int nt = 0; nt < 4; ++nt) acc[mt][nt] = v8f_zero();

  const int nT = K / 32;
  if (w == 0) {
    tdm_load_tile(offA[0], A + (size_t)row0 * K, K);
    tdm_load_tile(offB[0], B + (size_t)col0 * K, K);
  }

  for (int it = 0; it < nT; ++it) {
    const int cur = it & 1;
    if (w == 0) __builtin_amdgcn_s_wait_tensorcnt(0);
    __syncthreads();   // tile `it` landed; everyone done reading buf cur^1
    if (w == 0 && it + 1 < nT) {
      const int k0 = (it + 1) * 32;
      tdm_load_tile(offA[cur ^ 1], A + (size_t)row0 * K + k0, K);
      tdm_load_tile(offB[cur ^ 1], B + (size_t)col0 * K + k0, K);
    }
    const bf16* la = lA[cur];
    const bf16* lb = lB[cur];
#pragma unroll
    for (int mt = 0; mt < 2; ++mt) {
      const bf16* ap = &la[(waveR + mt * 16 + r) * 32 + g * 8];
      v16bf af = mk16(*(const v8bf*)ap, *(const v8bf*)(ap + 16));
#pragma unroll
      for (int nt = 0; nt < 4; ++nt) {
        const bf16* bp = &lb[(waveC + nt * 16 + r) * 32 + g * 8];
        v16bf bfr = mk16(*(const v8bf*)bp, *(const v8bf*)(bp + 16));
        acc[mt][nt] = __builtin_amdgcn_wmma_f32_16x16x32_bf16(
            false, af, false, bfr, (short)0, acc[mt][nt], false, false);
      }
    }
  }

#pragma unroll
  for (int mt = 0; mt < 2; ++mt)
#pragma unroll
    for (int nt = 0; nt < 4; ++nt) {
      size_t base = (size_t)(row0 + waveR + mt * 16 + 8 * g) * N +
                    (size_t)(col0 + waveC + nt * 16 + r);
#pragma unroll
      for (int v = 0; v < 8; ++v) C[base + (size_t)v * N] = acc[mt][nt][v];
    }
}

// ---------------------------------------------------------------- GEMM (sync fallback)
__global__ __launch_bounds__(256)
void gemm_bf16_wmma(const bf16* __restrict__ A, const bf16* __restrict__ B,
                    float* __restrict__ C, int M, int N, int K) {
  __shared__ __align__(16) bf16 lA[128 * 32];
  __shared__ __align__(16) bf16 lB[128 * 32];

  const int tid  = threadIdx.x;
  const int w    = tid >> 5;
  const int lane = tid & 31;
  const int g    = lane >> 4;
  const int r    = lane & 15;
  const int row0 = blockIdx.y * 128;
  const int col0 = blockIdx.x * 128;
  const int ldRow = tid >> 1;
  const int ldSeg = (tid & 1) * 16;
  const size_t aBase = (size_t)(row0 + ldRow) * K + ldSeg;
  const size_t bBase = (size_t)(col0 + ldRow) * K + ldSeg;
  const int waveR = (w >> 1) * 32;
  const int waveC = (w & 1) * 64;

  v8f acc[2][4];
#pragma unroll
  for (int mt = 0; mt < 2; ++mt)
#pragma unroll
    for (int nt = 0; nt < 4; ++nt) acc[mt][nt] = v8f_zero();

  for (int k0 = 0; k0 < K; k0 += 32) {
    v8bf av = *(const v8bf*)(A + aBase + k0);
    v8bf bv = *(const v8bf*)(B + bBase + k0);
    *(v8bf*)&lA[ldRow * 32 + ldSeg] = av;
    *(v8bf*)&lB[ldRow * 32 + ldSeg] = bv;
    if (k0 + 32 < K) {
      __builtin_prefetch(A + aBase + k0 + 32, 0, 1);
      __builtin_prefetch(B + bBase + k0 + 32, 0, 1);
    }
    __syncthreads();
#pragma unroll
    for (int mt = 0; mt < 2; ++mt) {
      const bf16* ap = &lA[(waveR + mt * 16 + r) * 32 + g * 8];
      v16bf af = mk16(*(const v8bf*)ap, *(const v8bf*)(ap + 16));
#pragma unroll
      for (int nt = 0; nt < 4; ++nt) {
        const bf16* bp = &lB[(waveC + nt * 16 + r) * 32 + g * 8];
        v16bf bfr = mk16(*(const v8bf*)bp, *(const v8bf*)(bp + 16));
        acc[mt][nt] = __builtin_amdgcn_wmma_f32_16x16x32_bf16(
            false, af, false, bfr, (short)0, acc[mt][nt], false, false);
      }
    }
    __syncthreads();
  }

#pragma unroll
  for (int mt = 0; mt < 2; ++mt)
#pragma unroll
    for (int nt = 0; nt < 4; ++nt) {
      size_t base = (size_t)(row0 + waveR + mt * 16 + 8 * g) * N +
                    (size_t)(col0 + waveC + nt * 16 + r);
#pragma unroll
      for (int v = 0; v < 8; ++v) C[base + (size_t)v * N] = acc[mt][nt][v];
    }
}

// ---------------------------------------------------------------- RMSNorm + RoPE + scatter
__global__ __launch_bounds__(128)
void rope_norm_scatter(const float* __restrict__ qkv, const int* __restrict__ pos,
                       const float* __restrict__ qw, const float* __restrict__ kw,
                       bf16* __restrict__ qt, bf16* __restrict__ kt,
                       bf16* __restrict__ vtr) {
  const int wave = blockIdx.x * 4 + (threadIdx.x >> 5);
  const int lane = threadIdx.x & 31;
  const int t = wave / (NHEAD + 2 * NKV);
  const int j = wave % (NHEAD + 2 * NKV);
  const int i2 = lane * 2;

  const float* row;
  if (j < 32)       row = qkv + (size_t)t * QKV_N + j * HD;
  else if (j < 36)  row = qkv + (size_t)t * QKV_N + QDIM + (j - 32) * HD;
  else              row = qkv + (size_t)t * QKV_N + QDIM + KVDIM + (j - 36) * HD;

  float x0 = row[i2], x1 = row[i2 + 1], x2 = row[i2 + 64], x3 = row[i2 + 65];
  const int p = pos[t];

  if (j < 36) {
    float ss = x0 * x0 + x1 * x1 + x2 * x2 + x3 * x3;
#pragma unroll
    for (int m = 1; m < 32; m <<= 1) ss += __shfl_xor(ss, m, 32);
    const float inv = rsqrtf(ss * (1.0f / 128.0f) + 1e-6f);
    const float* wv = (j < 32) ? qw : kw;
    x0 *= inv * wv[i2];      x1 *= inv * wv[i2 + 1];
    x2 *= inv * wv[i2 + 64]; x3 *= inv * wv[i2 + 65];

    const float nlt = -13.815510557964274f / 64.0f;   // -ln(1e6)/64
    float a0 = (float)p * __expf(nlt * (float)i2);
    float a1 = (float)p * __expf(nlt * (float)(i2 + 1));
    float s0, c0, s1, c1;
    __sincosf(a0, &s0, &c0);
    __sincosf(a1, &s1, &c1);
    float o0 = x0 * c0 - x2 * s0;
    float o1 = x1 * c1 - x3 * s1;
    float o2 = x2 * c0 + x0 * s0;
    float o3 = x3 * c1 + x1 * s1;

    bf16* dst = (j < 32) ? (qt + ((size_t)j * T_SEQ + t) * HD)
                         : (kt + ((size_t)(j - 32) * T_SEQ + p) * HD);
    v2bf lo; lo[0] = (bf16)o0; lo[1] = (bf16)o1;
    v2bf hi; hi[0] = (bf16)o2; hi[1] = (bf16)o3;
    *(v2bf*)(dst + i2)      = lo;
    *(v2bf*)(dst + i2 + 64) = hi;
  } else {
    bf16* dst = vtr + (size_t)(j - 36) * HD * T_SEQ;
    dst[(size_t)(i2)      * T_SEQ + p] = (bf16)x0;
    dst[(size_t)(i2 + 1)  * T_SEQ + p] = (bf16)x1;
    dst[(size_t)(i2 + 64) * T_SEQ + p] = (bf16)x2;
    dst[(size_t)(i2 + 65) * T_SEQ + p] = (bf16)x3;
  }
}

// ---------------------------------------------------------------- flash attention
__global__ __launch_bounds__(256)
void flash_attn(const bf16* __restrict__ qt, const bf16* __restrict__ kt,
                const bf16* __restrict__ vtr, bf16* __restrict__ y) {
  __shared__ __align__(16) bf16 pbuf[8][16 * 32];

  const int tid  = threadIdx.x;
  const int w    = tid >> 5;
  const int lane = tid & 31;
  const int g    = lane >> 4;
  const int r    = lane & 15;
  const int h    = blockIdx.y;
  const int kv   = h >> 3;
  const int t0   = blockIdx.x * 128 + w * 16;

  const bf16* qrow = qt + ((size_t)h * T_SEQ + t0 + r) * HD;
  v16bf qf[4];
#pragma unroll
  for (int kk = 0; kk < 4; ++kk)
    qf[kk] = mk16(*(const v8bf*)(qrow + kk * 32 + g * 8),
                  *(const v8bf*)(qrow + kk * 32 + 16 + g * 8));

  float m[8], l[8];
  v8f o[8];
#pragma unroll
  for (int v = 0; v < 8; ++v) { m[v] = -1e30f; l[v] = 0.0f; }
#pragma unroll
  for (int nt = 0; nt < 8; ++nt) o[nt] = v8f_zero();

  const float sscale = 0.08838834764831845f * 1.4426950408889634f;
  bf16* pb = pbuf[w];
  const int sEnd = t0 + 15;

  for (int s0 = 0; s0 <= sEnd; s0 += 32) {
    v8f sa[2];
    sa[0] = v8f_zero(); sa[1] = v8f_zero();
#pragma unroll
    for (int vt = 0; vt < 2; ++vt) {
      const bf16* krow = kt + ((size_t)kv * T_SEQ + s0 + vt * 16 + r) * HD;
#pragma unroll
      for (int kk = 0; kk < 4; ++kk) {
        v16bf bfr = mk16(*(const v8bf*)(krow + kk * 32 + g * 8),
                         *(const v8bf*)(krow + kk * 32 + 16 + g * 8));
        sa[vt] = __builtin_amdgcn_wmma_f32_16x16x32_bf16(
            false, qf[kk], false, bfr, (short)0, sa[vt], false, false);
      }
    }

    float sv[2][8];
    const bool diag = (s0 + 31 > t0);
#pragma unroll
    for (int vt = 0; vt < 2; ++vt)
#pragma unroll
      for (int v = 0; v < 8; ++v) {
        float val = sa[vt][v] * sscale;
        if (diag) {
          int s  = s0 + vt * 16 + r;
          int tq = t0 + v + 8 * g;
          if (s > tq) val = -1e30f;
        }
        sv[vt][v] = val;
      }

    float f[8];
#pragma unroll
    for (int v = 0; v < 8; ++v) {
      float mv = fmaxf(sv[0][v], sv[1][v]);
      mv = fmaxf(mv, __shfl_xor(mv, 1, 32));
      mv = fmaxf(mv, __shfl_xor(mv, 2, 32));
      mv = fmaxf(mv, __shfl_xor(mv, 4, 32));
      mv = fmaxf(mv, __shfl_xor(mv, 8, 32));
      float mn = fmaxf(m[v], mv);
      f[v] = exp2f(m[v] - mn);
      m[v] = mn;
      float p0 = exp2f(sv[0][v] - mn);
      float p1 = exp2f(sv[1][v] - mn);
      sv[0][v] = p0; sv[1][v] = p1;
      float rs = p0 + p1;
      rs += __shfl_xor(rs, 1, 32);
      rs += __shfl_xor(rs, 2, 32);
      rs += __shfl_xor(rs, 4, 32);
      rs += __shfl_xor(rs, 8, 32);
      l[v] = l[v] * f[v] + rs;
    }
#pragma unroll
    for (int nt = 0; nt < 8; ++nt)
#pragma unroll
      for (int v = 0; v < 8; ++v) o[nt][v] *= f[v];

#pragma unroll
    for (int vt = 0; vt < 2; ++vt)
#pragma unroll
      for (int v = 0; v < 8; ++v)
        pb[(v + 8 * g) * 32 + vt * 16 + r] = (bf16)sv[vt][v];
    asm volatile("s_wait_dscnt 0" ::: "memory");
    v16bf pf = mk16(*(const v8bf*)&pb[r * 32 + g * 8],
                    *(const v8bf*)&pb[r * 32 + 16 + g * 8]);

#pragma unroll
    for (int nt = 0; nt < 8; ++nt) {
      const bf16* vrow = vtr + ((size_t)kv * HD + nt * 16 + r) * T_SEQ + s0 + g * 8;
      v16bf bfr = mk16(*(const v8bf*)vrow, *(const v8bf*)(vrow + 16));
      o[nt] = __builtin_amdgcn_wmma_f32_16x16x32_bf16(
          false, pf, false, bfr, (short)0, o[nt], false, false);
    }
  }

  float rl[8];
#pragma unroll
  for (int v = 0; v < 8; ++v) rl[v] = 1.0f / l[v];
#pragma unroll
  for (int nt = 0; nt < 8; ++nt)
#pragma unroll
    for (int v = 0; v < 8; ++v)
      y[(size_t)(t0 + v + 8 * g) * QDIM + h * HD + nt * 16 + r] =
          (bf16)(o[nt][v] * rl[v]);
}

// ---------------------------------------------------------------- launch
extern "C" void kernel_launch(void* const* d_in, const int* in_sizes, int n_in,
                              void* d_out, int out_size, void* d_ws, size_t ws_size,
                              hipStream_t stream) {
  const float* x   = (const float*)d_in[0];
  const int*   pos = (const int*)d_in[1];
  const float* Wq  = (const float*)d_in[2];
  const float* Wk  = (const float*)d_in[3];
  const float* Wv  = (const float*)d_in[4];
  const float* Wo  = (const float*)d_in[5];
  const float* qnw = (const float*)d_in[6];
  const float* knw = (const float*)d_in[7];
  float* out = (float*)d_out;

  char* p = (char*)d_ws;
  auto alloc = [&](size_t bytes) -> char* {
    char* ret = p;
    p += (bytes + 255) & ~(size_t)255;
    return ret;
  };
  bf16*  xb   = (bf16*)alloc((size_t)T_SEQ * C_EMB * 2);
  bf16*  wqkv = (bf16*)alloc((size_t)QKV_N * C_EMB * 2);
  bf16*  wo   = (bf16*)alloc((size_t)C_EMB * QDIM * 2);
  float* qkv  = (float*)alloc((size_t)T_SEQ * QKV_N * 4);
  bf16*  qt   = (bf16*)alloc((size_t)NHEAD * T_SEQ * HD * 2);
  bf16*  kt   = (bf16*)alloc((size_t)NKV * T_SEQ * HD * 2);
  bf16*  vtr  = (bf16*)alloc((size_t)NKV * HD * T_SEQ * 2);
  bf16*  y    = (bf16*)alloc((size_t)T_SEQ * QDIM * 2);

  auto cvt = [&](const float* s, bf16* d, size_t n) {
    int n4 = (int)(n / 4);
    cvt_f32_bf16<<<(n4 + 255) / 256, 256, 0, stream>>>(s, d, n4);
  };
  cvt(x,  xb,   (size_t)T_SEQ * C_EMB);
  cvt(Wq, wqkv, (size_t)QDIM * C_EMB);
  cvt(Wk, wqkv + (size_t)QDIM * C_EMB,  (size_t)KVDIM * C_EMB);
  cvt(Wv, wqkv + (size_t)(QDIM + KVDIM) * C_EMB, (size_t)KVDIM * C_EMB);
  cvt(Wo, wo,   (size_t)C_EMB * QDIM);

  // fused QKV projection (TDM-fed WMMA GEMM)
  gemm_bf16_wmma_tdm<<<dim3(QKV_N / 128, T_SEQ / 128), 256, 0, stream>>>(
      xb, wqkv, qkv, T_SEQ, QKV_N, C_EMB);

  rope_norm_scatter<<<(T_SEQ * (NHEAD + 2 * NKV)) / 4, 128, 0, stream>>>(
      qkv, pos, qnw, knw, qt, kt, vtr);

  flash_attn<<<dim3(T_SEQ / 128, NHEAD), 256, 0, stream>>>(qt, kt, vtr, y);

  // output projection (TDM-fed WMMA GEMM)
  gemm_bf16_wmma_tdm<<<dim3(C_EMB / 128, T_SEQ / 128), 256, 0, stream>>>(
      y, wo, out, T_SEQ, C_EMB, QDIM);
}
